// MultiHeadAttention_44358422233370
// MI455X (gfx1250) — compile-verified, matrix operand order checked by reading, not run on hardware
//
#include <hip/hip_runtime.h>
#include <hip/hip_bf16.h>

typedef unsigned short u16;
typedef __bf16 bf16_t;
typedef bf16_t v16bf __attribute__((ext_vector_type(16)));
typedef float  v8f   __attribute__((ext_vector_type(8)));
typedef unsigned int uv4 __attribute__((ext_vector_type(4)));
typedef int iv8 __attribute__((ext_vector_type(8)));
typedef int iv4 __attribute__((ext_vector_type(4)));

union FragAB  { v16bf v; uint4 q[2]; };
union Pack4   { uint2 u; u16 s[4]; };

__device__ __forceinline__ u16 f2bf_bits(float f) {
  union { float f; unsigned u; } v; v.f = f;
  unsigned r = (v.u + 0x7FFFu + ((v.u >> 16) & 1u)) >> 16;  // RNE
  return (u16)r;
}

// Build a 16x32 A-fragment (or mirrored B-fragment) from two contiguous
// 16-byte K-chunks per lane (ISA 16-bit A layout: lane=M, chunks at
// K = 8*half .. +7 and K = 16+8*half .. +7).
__device__ __forceinline__ v16bf load_frag(const u16* c0, const u16* c1) {
  FragAB f;
  f.q[0] = *reinterpret_cast<const uint4*>(c0);
  f.q[1] = *reinterpret_cast<const uint4*>(c1);
  return f.v;
}

__device__ __forceinline__ v16bf ones_frag() {
  FragAB f;
  const uint4 o = {0x3F803F80u, 0x3F803F80u, 0x3F803F80u, 0x3F803F80u};
  f.q[0] = o; f.q[1] = o;                   // bf16 1.0 broadcast
  return f.v;
}

#define BM 128
#define BN 128
#define BK 64
#define LDSS 72   // 144-byte row stride -> conflict-free 16B fragment loads

// C = A @ W^T + bias.  A: [M x Kdim] (fp32 or bf16), W: [N x Kdim] fp32.
// mode 0: store bf16 to [B,H,S,D] (val*scale);  mode 1: store fp32 row-major.
template <bool A_BF16>
__global__ void __launch_bounds__(256) wmma_gemm_kernel(
    const void* __restrict__ A_, const float* __restrict__ W,
    const float* __restrict__ bias, void* __restrict__ out,
    int Kdim, int mode, float scale) {
  __shared__ __align__(16) u16 sA[BM * LDSS];
  __shared__ __align__(16) u16 sB[BN * LDSS];

  const int tid = threadIdx.x, lane = tid & 31;
  const int hf = lane >> 4, l16 = lane & 15;
  const int wave = tid >> 5, wm = wave >> 2, wn = wave & 3;
  const int rowbase = blockIdx.y * BM;
  const int colbase = blockIdx.x * BN;

  const v8f vzero = {0.f, 0.f, 0.f, 0.f, 0.f, 0.f, 0.f, 0.f};
  v8f acc[4][2];
#pragma unroll
  for (int i = 0; i < 4; ++i)
#pragma unroll
    for (int j = 0; j < 2; ++j) acc[i][j] = vzero;

  for (int k0 = 0; k0 < Kdim; k0 += BK) {
    __syncthreads();
    if (A_BF16) {
      const u16* A = (const u16*)A_;
#pragma unroll
      for (int p = 0; p < 4; ++p) {                  // 1024 8-elem chunks
        int c = tid + p * 256, r = c >> 3, c8 = c & 7;
        uint4 d = *reinterpret_cast<const uint4*>(
            A + (size_t)(rowbase + r) * Kdim + k0 + c8 * 8);
        *reinterpret_cast<uint4*>(&sA[r * LDSS + c8 * 8]) = d;
      }
    } else {
      const float* A = (const float*)A_;
#pragma unroll
      for (int p = 0; p < 8; ++p) {                  // 2048 float4 chunks
        int f = tid + p * 256, r = f >> 4, c4 = f & 15;
        float4 d = *reinterpret_cast<const float4*>(
            A + (size_t)(rowbase + r) * Kdim + k0 + c4 * 4);
        Pack4 pk;
        pk.s[0] = f2bf_bits(d.x); pk.s[1] = f2bf_bits(d.y);
        pk.s[2] = f2bf_bits(d.z); pk.s[3] = f2bf_bits(d.w);
        *reinterpret_cast<uint2*>(&sA[r * LDSS + c4 * 4]) = pk.u;
      }
    }
#pragma unroll
    for (int p = 0; p < 8; ++p) {                    // stage W (fp32->bf16)
      int f = tid + p * 256, r = f >> 4, c4 = f & 15;
      float4 d = *reinterpret_cast<const float4*>(
          W + (size_t)(colbase + r) * Kdim + k0 + c4 * 4);
      Pack4 pk;
      pk.s[0] = f2bf_bits(d.x); pk.s[1] = f2bf_bits(d.y);
      pk.s[2] = f2bf_bits(d.z); pk.s[3] = f2bf_bits(d.w);
      *reinterpret_cast<uint2*>(&sB[r * LDSS + c4 * 4]) = pk.u;
    }
    __syncthreads();

#pragma unroll
    for (int ks = 0; ks < 2; ++ks) {                 // two WMMA K-substeps
      v16bf af[4], bfr[2];
#pragma unroll
      for (int i = 0; i < 4; ++i) {
        const u16* rp = &sA[(wm * 64 + i * 16 + l16) * LDSS + 32 * ks];
        af[i] = load_frag(rp + 8 * hf, rp + 16 + 8 * hf);
      }
#pragma unroll
      for (int j = 0; j < 2; ++j) {
        const u16* rp = &sB[(wn * 32 + j * 16 + l16) * LDSS + 32 * ks];
        bfr[j] = load_frag(rp + 8 * hf, rp + 16 + 8 * hf);
      }
#pragma unroll
      for (int i = 0; i < 4; ++i)
#pragma unroll
        for (int j = 0; j < 2; ++j)
          acc[i][j] = __builtin_amdgcn_wmma_f32_16x16x32_bf16(
              false, af[i], false, bfr[j], (short)0, acc[i][j], false, false);
    }
  }

#pragma unroll
  for (int i = 0; i < 4; ++i)
#pragma unroll
    for (int j = 0; j < 2; ++j)
#pragma unroll
      for (int r = 0; r < 8; ++r) {
        int m = rowbase + wm * 64 + i * 16 + r + 8 * hf;
        int n = colbase + wn * 32 + j * 16 + l16;
        float v = (acc[i][j][r] + bias[n]) * scale;
        if (mode == 0) {
          int b = m >> 11, s = m & 2047, h = n >> 6, d = n & 63;
          ((u16*)out)[(((size_t)(b * 16 + h) * 2048 + s) * 64) + d] =
              f2bf_bits(v);
        } else {
          ((float*)out)[(size_t)m * 1024 + n] = v;
        }
      }
}

#define KST 72  // 144-byte row stride for attention LDS tiles

// Flash attention, causal.  Q,K,V bf16 [B,H,S,D] (Q pre-scaled); O bf16 [B,S,E].
// K tiles staged LDS via the Tensor Data Mover (pad_enable gives the 144B
// row stride); V tiles staged manually with a transpose + pi-permutation of
// the contraction index so P stores pack into b64 DS stores.
__global__ void __launch_bounds__(256) attn_fwd_kernel(
    const u16* __restrict__ Q, const u16* __restrict__ K,
    const u16* __restrict__ V, u16* __restrict__ O) {
  __shared__ __align__(16) u16 sK[64 * KST];
  __shared__ __align__(16) u16 sVt[64 * KST];
  __shared__ __align__(16) u16 sP[8 * 16 * KST];

  const int tid = threadIdx.x, lane = tid & 31;
  const int hf = lane >> 4, l16 = lane & 15, wave = tid >> 5;
  const int bh = blockIdx.y, b = bh >> 4, h = bh & 15;
  const int qtile = blockIdx.x, qbase = qtile * 128;
  const size_t headoff = (size_t)bh * 2048 * 64;
  const u16* Qh = Q + headoff;
  const u16* Kh = K + headoff;
  const u16* Vh = V + headoff;

  // Q fragments held in registers for the whole kernel (A: lane = M row)
  const int qrow = qbase + wave * 16 + l16;
  v16bf qf[2];
#pragma unroll
  for (int j = 0; j < 2; ++j) {
    const u16* rp = Qh + (size_t)qrow * 64 + 32 * j;
    qf[j] = load_frag(rp + 8 * hf, rp + 16 + 8 * hf);
  }
  const v16bf onef = ones_frag();

  const v8f vzero = {0.f, 0.f, 0.f, 0.f, 0.f, 0.f, 0.f, 0.f};
  float m_s[8];
  v8f oacc[4], lacc = vzero;
#pragma unroll
  for (int r = 0; r < 8; ++r) m_s[r] = -1e30f;
#pragma unroll
  for (int d = 0; d < 4; ++d) oacc[d] = vzero;

  u16* sPw = &sP[wave * 16 * KST];
  const int ntiles = 2 * (qtile + 1);  // causal limit, uniform across block

  for (int t = 0; t < ntiles; ++t) {
    const int tbase = t * 64;
    __syncthreads();

    // ---- K tile via Tensor Data Mover (wave 0 issues; EXEC ignored) ----
    if (wave == 0) {
      unsigned long long ga = (unsigned long long)(Kh + (size_t)tbase * 64);
      unsigned la = (unsigned)(unsigned long long)&sK[0];
      uv4 g0;
      g0[0] = 1u;                                   // count=1 valid user D#
      g0[1] = la;                                   // lds_addr
      g0[2] = (unsigned)(ga & 0xFFFFFFFFull);       // global_addr[31:0]
      g0[3] = (unsigned)((ga >> 32) & 0x1FFFFFFull) // global_addr[56:32]
              | (2u << 30);                         // type = image
      iv8 g1;
      g1[0] = (int)((1u << 16)       // data_size = 2 bytes
                    | (1u << 20)     // pad_enable
                    | (4u << 22)     // pad_interval: every 32 DWORDs
                    | (3u << 25));   // pad_amount: 4 DWORDs (-> 144B rows)
      g1[1] = (int)(64u << 16);      // tensor_dim0 = 64
      g1[2] = (int)(2048u << 16);    // tensor_dim1 = 2048
      g1[3] = (int)(64u << 16);      // tile_dim0 = 64
      g1[4] = (int)64u;              // tile_dim1 = 64, tile_dim2 = 0
      g1[5] = (int)64u;              // tensor_dim0_stride = 64
      g1[6] = 0;
      g1[7] = 0;
      const iv4 z4 = {0, 0, 0, 0};
      const iv8 z8 = {0, 0, 0, 0, 0, 0, 0, 0};
      __builtin_amdgcn_tensor_load_to_lds(g0, g1, z4, z4, z8, 0);
    }

    // ---- V tile, transposed + pi-permuted columns: pi(t) = (t&15)*4+(t>>4)
#pragma unroll
    for (int p = 0; p < 4; ++p) {
      int c = tid + p * 256, r = c >> 4, c4 = c & 15;
      const int pr = (r & 15) * 4 + (r >> 4);
      Pack4 pk;
      pk.u = *reinterpret_cast<const uint2*>(
          Vh + (size_t)(tbase + r) * 64 + c4 * 4);
#pragma unroll
      for (int e = 0; e < 4; ++e) sVt[(c4 * 4 + e) * KST + pr] = pk.s[e];
    }
    if (wave == 0) __builtin_amdgcn_s_wait_tensorcnt(0);
    __syncthreads();

    if (t + 1 < ntiles)  // global_prefetch_b8, pull next V tile toward WGP
      __builtin_prefetch(Vh + (size_t)(tbase + 64 + (tid >> 2)) * 64, 0, 3);

    // scores(16x64) = Q(16x64) @ K^T
    v8f sc[4];
#pragma unroll
    for (int nt = 0; nt < 4; ++nt) {
      v8f c = vzero;
#pragma unroll
      for (int j = 0; j < 2; ++j) {
        const u16* rp = &sK[(nt * 16 + l16) * KST + 32 * j];
        v16bf bfr = load_frag(rp + 8 * hf, rp + 16 + 8 * hf);
        c = __builtin_amdgcn_wmma_f32_16x16x32_bf16(
            false, qf[j], false, bfr, (short)0, c, false, false);
      }
      sc[nt] = c;
    }

    // causal mask + online softmax (rows = r + 8*hf; cols across 16 lanes)
    float corr[8];
#pragma unroll
    for (int r = 0; r < 8; ++r) {
      const int srow = qbase + wave * 16 + r + 8 * hf;
      float tmax = -1e30f;
#pragma unroll
      for (int f = 0; f < 4; ++f) {
        const int col = tbase + f * 16 + l16;
        float v = sc[f][r];
        v = (col <= srow) ? v : -1e30f;
        sc[f][r] = v;
        tmax = fmaxf(tmax, v);
      }
      tmax = fmaxf(tmax, __shfl_xor(tmax, 1, 32));
      tmax = fmaxf(tmax, __shfl_xor(tmax, 2, 32));
      tmax = fmaxf(tmax, __shfl_xor(tmax, 4, 32));
      tmax = fmaxf(tmax, __shfl_xor(tmax, 8, 32));
      const float mnew = fmaxf(m_s[r], tmax);
      corr[r] = __expf(m_s[r] - mnew);
      m_s[r] = mnew;
      Pack4 pk;                        // P row r+8hf, cols pi-packed: b64 store
#pragma unroll
      for (int f = 0; f < 4; ++f) pk.s[f] = f2bf_bits(__expf(sc[f][r] - mnew));
      *reinterpret_cast<uint2*>(&sPw[(r + 8 * hf) * KST + l16 * 4]) = pk.u;
    }
    asm volatile("s_wait_dscnt 0" ::: "memory");  // P visible to own wave

    // rescale accumulators, then oacc += P@V and lacc += P@ones (row sums)
#pragma unroll
    for (int r = 0; r < 8; ++r) {
      lacc[r] *= corr[r];
#pragma unroll
      for (int dt = 0; dt < 4; ++dt) oacc[dt][r] *= corr[r];
    }
    v16bf pf[2];
#pragma unroll
    for (int j = 0; j < 2; ++j) {
      const u16* ap = &sPw[l16 * KST + 32 * j];
      pf[j] = load_frag(ap + 8 * hf, ap + 16 + 8 * hf);
    }
#pragma unroll
    for (int dt = 0; dt < 4; ++dt) {
      v8f c = oacc[dt];
#pragma unroll
      for (int j = 0; j < 2; ++j) {
        const u16* bp = &sVt[(dt * 16 + l16) * KST + 32 * j];
        v16bf bfr = load_frag(bp + 8 * hf, bp + 16 + 8 * hf);
        c = __builtin_amdgcn_wmma_f32_16x16x32_bf16(
            false, pf[j], false, bfr, (short)0, c, false, false);
      }
      oacc[dt] = c;
    }
#pragma unroll
    for (int j = 0; j < 2; ++j)
      lacc = __builtin_amdgcn_wmma_f32_16x16x32_bf16(
          false, pf[j], false, onef, (short)0, lacc, false, false);
  }

#pragma unroll
  for (int r = 0; r < 8; ++r) {
    const float inv = 1.0f / lacc[r];  // every lane holds its row's sum
    const int s = qbase + wave * 16 + r + 8 * hf;
#pragma unroll
    for (int dt = 0; dt < 4; ++dt) {
      const int e = h * 64 + dt * 16 + l16;
      O[((size_t)(b * 2048 + s)) * 1024 + e] = f2bf_bits(oacc[dt][r] * inv);
    }
  }
}

extern "C" void kernel_launch(void* const* d_in, const int* in_sizes, int n_in,
                              void* d_out, int out_size, void* d_ws,
                              size_t ws_size, hipStream_t stream) {
  const float* x  = (const float*)d_in[0];
  const float* Wq = (const float*)d_in[1];
  const float* bq = (const float*)d_in[2];
  const float* Wk = (const float*)d_in[3];
  const float* bk = (const float*)d_in[4];
  const float* Wv = (const float*)d_in[5];
  const float* bv = (const float*)d_in[6];
  const float* Wo = (const float*)d_in[7];
  const float* bo = (const float*)d_in[8];

  u16* Qb = (u16*)d_ws;                      // [B,H,S,D] bf16, pre-scaled
  u16* Kb = Qb + (size_t)4096 * 1024;        // [B,H,S,D] bf16
  u16* Vb = Kb + (size_t)4096 * 1024;        // [B,H,S,D] bf16
  u16* AO = Vb + (size_t)4096 * 1024;        // [B,S,E]   bf16

  dim3 ggrid(1024 / BN, 4096 / BM), blk(256);
  wmma_gemm_kernel<false><<<ggrid, blk, 0, stream>>>(x, Wq, bq, Qb, 1024, 0, 0.125f);
  wmma_gemm_kernel<false><<<ggrid, blk, 0, stream>>>(x, Wk, bk, Kb, 1024, 0, 1.0f);
  wmma_gemm_kernel<false><<<ggrid, blk, 0, stream>>>(x, Wv, bv, Vb, 1024, 0, 1.0f);
  attn_fwd_kernel<<<dim3(16, 32), blk, 0, stream>>>(Qb, Kb, Vb, AO);
  wmma_gemm_kernel<true><<<ggrid, blk, 0, stream>>>(AO, Wo, bo, d_out, 1024, 1, 1.0f);
}
